// LayoutReaderSelfAttention_42726334661206
// MI455X (gfx1250) — matrix operand order repaired and checked by reading, not compiled, not run on hardware
//
#include <hip/hip_runtime.h>
#include <hip/hip_bf16.h>

typedef __attribute__((ext_vector_type(16))) __bf16 v16bf;
typedef __attribute__((ext_vector_type(8)))  __bf16 v8bf;
typedef __attribute__((ext_vector_type(8)))  float  v8f;

#define S_LEN 2048
#define HDIM  1024
#define NH    16
#define HD    64
#define BATCH 2
#define MTOT  (BATCH * S_LEN)   // 4096 rows for the projection GEMMs

static __device__ __forceinline__ v16bf concat8(v8bf lo, v8bf hi) {
  return __builtin_shufflevector(lo, hi, 0,1,2,3,4,5,6,7,8,9,10,11,12,13,14,15);
}

static __device__ __forceinline__ v8f wmma_bf16(v16bf a, v16bf b, v8f c) {
  return __builtin_amdgcn_wmma_f32_16x16x32_bf16(false, a, false, b, (short)0, c,
                                                 false, false);
}

// CDNA5 async memory->LDS copy (ASYNCcnt-tracked, no VGPR round trip).
// VDST VGPR holds the LDS byte address: the low 32 bits of a flat pointer into
// the LDS aperture are exactly the LDS address (ISA 10.2 aperture mapping).
static __device__ __forceinline__ void async_load_lds_b128(void* lds_ptr,
                                                           const void* gptr) {
  unsigned int lds_addr = (unsigned int)(size_t)lds_ptr;
  asm volatile("global_load_async_to_lds_b128 %0, %1, off"
               :: "v"(lds_addr), "v"(gptr)
               : "memory");
}
static __device__ __forceinline__ void wait_async_lds() {
  asm volatile("s_wait_asynccnt 0x0" ::: "memory");
}

// A fragment (16x32, bf16) per CDNA5 ISA layout:
// lane = 16*half + m; VGPRs0-3 hold K = half*8+0..7, VGPRs4-7 hold K = 16+half*8+0..7.
// Source is row-major with K contiguous -> two 16B chunks per lane.
template <typename P>
static __device__ __forceinline__ v16bf load_a(const P* base, int rowstride,
                                               int row, int kk, int lane) {
  const P* r = base + (size_t)(row + (lane & 15)) * rowstride;
  int half = lane >> 4;
  v8bf lo = *(const v8bf*)(r + kk + half * 8);
  v8bf hi = *(const v8bf*)(r + kk + 16 + half * 8);
  return concat8(lo, hi);
}

// B fragment (32x16, bf16): lane = 16*half + n; VGPRs0-3 hold K = half*8+0..7, VGPRs4-7 hold K = 16+half*8+0..7.
// Source row n holds its K values contiguously -> two 16B chunks per lane.
template <typename P>
static __device__ __forceinline__ v16bf load_b(const P* base, int rowstride,
                                               int nrow, int kk, int lane) {
  const P* r = base + (size_t)(nrow + (lane & 15)) * rowstride;
  int half = lane >> 4;
  v8bf lo = *(const v8bf*)(r + kk + half * 8);
  v8bf hi = *(const v8bf*)(r + kk + 16 + half * 8);
  return concat8(lo, hi);
}

__global__ void cvt_f32_bf16(const float* __restrict__ in,
                             __bf16* __restrict__ out, int n) {
  int i = blockIdx.x * blockDim.x + threadIdx.x;
  int stride = gridDim.x * blockDim.x;
  for (; i < n; i += stride) out[i] = (__bf16)in[i];
}

// Y[m,n] = sum_k X[m,k]*W[n,k]  (+bias[n]) * scale, stored bf16.
// out_mode 0: out[((b*NH+h)*S + s)*HD + d]   (Q, K)
// out_mode 1: out[((b*NH+h)*HD + d)*S + s]   (V transposed for P@V)
__global__ __launch_bounds__(256)
void gemm_qkv(const __bf16* __restrict__ X, const __bf16* __restrict__ W,
              const float* __restrict__ bias, __bf16* __restrict__ out,
              int out_mode, float scale) {
  int lane = threadIdx.x & 31;
  int wave = threadIdx.x >> 5;             // 8 waves
  int mbase = blockIdx.x * 128 + (wave >> 2) * 64;   // 2 wave-rows of 64
  int nbase = blockIdx.y * 128 + (wave & 3) * 32;    // 4 wave-cols of 32

  v8f acc[4][2];
#pragma unroll
  for (int t = 0; t < 4; ++t)
#pragma unroll
    for (int u = 0; u < 2; ++u)
#pragma unroll
      for (int i = 0; i < 8; ++i) acc[t][u][i] = 0.0f;

  for (int k0 = 0; k0 < HDIM; k0 += 32) {
    v16bf a[4], b[2];
#pragma unroll
    for (int t = 0; t < 4; ++t) a[t] = load_a(X, HDIM, mbase + t * 16, k0, lane);
#pragma unroll
    for (int u = 0; u < 2; ++u) b[u] = load_b(W, HDIM, nbase + u * 16, k0, lane);
#pragma unroll
    for (int t = 0; t < 4; ++t)
#pragma unroll
      for (int u = 0; u < 2; ++u)
        acc[t][u] = wmma_bf16(a[t], b[u], acc[t][u]);
  }

  int mloc = (lane >> 4) * 8;
  int nloc = lane & 15;
#pragma unroll
  for (int t = 0; t < 4; ++t) {
#pragma unroll
    for (int u = 0; u < 2; ++u) {
      int n = nbase + u * 16 + nloc;
      float bval = bias ? bias[n] : 0.0f;
      int h = n >> 6, d = n & 63;
#pragma unroll
      for (int r = 0; r < 8; ++r) {
        int m = mbase + t * 16 + mloc + r;
        int bb = m >> 11, s = m & 2047;
        float v = (acc[t][u][r] + bval) * scale;
        size_t idx = (out_mode == 0)
            ? ((size_t)(bb * NH + h) * S_LEN + s) * HD + d
            : ((size_t)(bb * NH + h) * HD + d) * S_LEN + s;
        out[idx] = (__bf16)v;
      }
    }
  }
}

// One workgroup = 128 q rows of one (b,h). kv tiles of 64, two-pass softmax.
__global__ __launch_bounds__(256)
void attn(const __bf16* __restrict__ Qb, const __bf16* __restrict__ Kb,
          const __bf16* __restrict__ Vt, const float* __restrict__ mask,
          float* __restrict__ out) {
  __shared__ __bf16 Qs[128 * HD];   // 16 KB
  __shared__ __bf16 Ks[64 * HD];    //  8 KB
  __shared__ float  Ss[128 * 64];   // 32 KB (P bf16 overlays row starts in phase 2)

  int tid = threadIdx.x;
  int lane = tid & 31, wave = tid >> 5;
  int qblk = blockIdx.x;   // 0..15
  int h    = blockIdx.y;   // 0..15
  int b    = blockIdx.z;   // 0..1

  const __bf16* Qg = Qb + ((size_t)(b * NH + h) * S_LEN + qblk * 128) * HD;
  const __bf16* Kg = Kb + (size_t)(b * NH + h) * S_LEN * HD;
  const __bf16* Vg = Vt + (size_t)(b * NH + h) * HD * S_LEN;
  const float*  mk = mask + (size_t)b * S_LEN;

  // stage Q tile (8192 elems = 16 KB) with async memory->LDS copies
#pragma unroll
  for (int i = 0; i < 4; ++i) {
    int e = (tid * 4 + i) * 8;   // 8 bf16 = 16 B per b128
    async_load_lds_b128(Qs + e, Qg + e);
  }
  wait_async_lds();

  float m_i = -1e30f, l_i = 0.0f;
  int qrow = tid;  // valid for tid < 128

  // ---------- pass 1: row max & sum of exp ----------
  for (int kv0 = 0; kv0 < S_LEN; kv0 += 64) {
    __syncthreads();
    {
      int e = tid * 16;  // 4096 elems / 256 threads, 32 B each
      async_load_lds_b128(Ks + e, Kg + (size_t)kv0 * HD + e);
      async_load_lds_b128(Ks + e + 8, Kg + (size_t)kv0 * HD + e + 8);
      if (kv0 + 64 < S_LEN)
        __builtin_prefetch(Kg + (size_t)(kv0 + 64) * HD + e, 0, 1);
      wait_async_lds();
    }
    __syncthreads();

    v8f sacc[4];
#pragma unroll
    for (int u = 0; u < 4; ++u)
#pragma unroll
      for (int i = 0; i < 8; ++i) sacc[u][i] = 0.0f;
#pragma unroll
    for (int ks = 0; ks < 2; ++ks) {
      v16bf a = load_a(Qs, HD, wave * 16, ks * 32, lane);
#pragma unroll
      for (int u = 0; u < 4; ++u) {
        v16bf bb = load_b(Ks, HD, u * 16, ks * 32, lane);
        sacc[u] = wmma_bf16(a, bb, sacc[u]);
      }
    }
    int mloc = (lane >> 4) * 8, nloc = lane & 15;
#pragma unroll
    for (int u = 0; u < 4; ++u)
#pragma unroll
      for (int r = 0; r < 8; ++r)
        Ss[(wave * 16 + mloc + r) * 64 + u * 16 + nloc] = sacc[u][r];
    __syncthreads();

    if (tid < 128) {
      float tmax = -1e30f;
      for (int j = 0; j < 64; ++j)
        tmax = fmaxf(tmax, Ss[qrow * 64 + j] + mk[kv0 + j]);
      float mnew = fmaxf(m_i, tmax);
      float lsum = 0.0f;
      for (int j = 0; j < 64; ++j)
        lsum += __expf(Ss[qrow * 64 + j] + mk[kv0 + j] - mnew);
      l_i = l_i * __expf(m_i - mnew) + lsum;
      m_i = mnew;
    }
  }
  float inv_l = (tid < 128) ? 1.0f / l_i : 0.0f;

  // ---------- pass 2: P = exp(s-m)/l, ctx += P @ V ----------
  v8f cacc[4];
#pragma unroll
  for (int u = 0; u < 4; ++u)
#pragma unroll
    for (int i = 0; i < 8; ++i) cacc[u][i] = 0.0f;

  for (int kv0 = 0; kv0 < S_LEN; kv0 += 64) {
    __syncthreads();
    {
      int e = tid * 16;
      async_load_lds_b128(Ks + e, Kg + (size_t)kv0 * HD + e);
      async_load_lds_b128(Ks + e + 8, Kg + (size_t)kv0 * HD + e + 8);
      if (kv0 + 64 < S_LEN)
        __builtin_prefetch(Kg + (size_t)(kv0 + 64) * HD + e, 0, 1);
      wait_async_lds();
    }
    __syncthreads();

    v8f sacc[4];
#pragma unroll
    for (int u = 0; u < 4; ++u)
#pragma unroll
      for (int i = 0; i < 8; ++i) sacc[u][i] = 0.0f;
#pragma unroll
    for (int ks = 0; ks < 2; ++ks) {
      v16bf a = load_a(Qs, HD, wave * 16, ks * 32, lane);
#pragma unroll
      for (int u = 0; u < 4; ++u) {
        v16bf bb = load_b(Ks, HD, u * 16, ks * 32, lane);
        sacc[u] = wmma_bf16(a, bb, sacc[u]);
      }
    }
    int mloc = (lane >> 4) * 8, nloc = lane & 15;
#pragma unroll
    for (int u = 0; u < 4; ++u)
#pragma unroll
      for (int r = 0; r < 8; ++r)
        Ss[(wave * 16 + mloc + r) * 64 + u * 16 + nloc] = sacc[u][r];
    __syncthreads();

    if (tid < 128) {
      // convert row to bf16 P in place: bf16 row r lives at byte offset r*256
      __bf16* prow = (__bf16*)(Ss + qrow * 64);
      for (int j = 0; j < 64; ++j) {
        float v = Ss[qrow * 64 + j] + mk[kv0 + j];
        prow[j] = (__bf16)(__expf(v - m_i) * inv_l);
      }
    }
    __syncthreads();

    const __bf16* Pbase = (const __bf16*)Ss;  // row stride 128 bf16 elems
#pragma unroll
    for (int ks = 0; ks < 2; ++ks) {
      v16bf a = load_a(Pbase, 128, wave * 16, ks * 32, lane);
#pragma unroll
      for (int u = 0; u < 4; ++u) {
        v16bf bb = load_b(Vg, S_LEN, u * 16, kv0 + ks * 32, lane);
        cacc[u] = wmma_bf16(a, bb, cacc[u]);
      }
    }
  }

  int mloc = (lane >> 4) * 8, nloc = lane & 15;
#pragma unroll
  for (int u = 0; u < 4; ++u) {
    int d = u * 16 + nloc;
#pragma unroll
    for (int r = 0; r < 8; ++r) {
      int q = qblk * 128 + wave * 16 + mloc + r;
      out[((size_t)b * S_LEN + q) * HDIM + h * HD + d] = cacc[u][r];
    }
  }
}

extern "C" void kernel_launch(void* const* d_in, const int* in_sizes, int n_in,
                              void* d_out, int out_size, void* d_ws, size_t ws_size,
                              hipStream_t stream) {
  const float* X    = (const float*)d_in[0];
  const float* mask = (const float*)d_in[1];
  const float* Wq   = (const float*)d_in[2];
  const float* bq   = (const float*)d_in[3];
  const float* Wk   = (const float*)d_in[4];
  const float* Wv   = (const float*)d_in[5];
  const float* bv   = (const float*)d_in[6];
  float* out = (float*)d_out;

  char* ws = (char*)d_ws;
  const size_t XB   = (size_t)MTOT * HDIM * 2;   // 8 MB
  const size_t WB   = (size_t)HDIM * HDIM * 2;   // 2 MB
  __bf16* Xb  = (__bf16*)(ws);
  __bf16* Wqb = (__bf16*)(ws + XB);
  __bf16* Wkb = (__bf16*)(ws + XB + WB);
  __bf16* Wvb = (__bf16*)(ws + XB + 2 * WB);
  __bf16* Qb  = (__bf16*)(ws + XB + 3 * WB);
  __bf16* Kb  = (__bf16*)(ws + 2 * XB + 3 * WB);
  __bf16* Vt  = (__bf16*)(ws + 3 * XB + 3 * WB);

  cvt_f32_bf16<<<4096, 256, 0, stream>>>(X,  Xb,  MTOT * HDIM);
  cvt_f32_bf16<<<1024, 256, 0, stream>>>(Wq, Wqb, HDIM * HDIM);
  cvt_f32_bf16<<<1024, 256, 0, stream>>>(Wk, Wkb, HDIM * HDIM);
  cvt_f32_bf16<<<1024, 256, 0, stream>>>(Wv, Wvb, HDIM * HDIM);

  dim3 g(MTOT / 128, HDIM / 128);  // 32 x 8
  gemm_qkv<<<g, 256, 0, stream>>>(Xb, Wqb, bq,      Qb, 0, 0.125f);  // Q, pre-scaled
  gemm_qkv<<<g, 256, 0, stream>>>(Xb, Wkb, nullptr, Kb, 0, 1.0f);    // K, no bias
  gemm_qkv<<<g, 256, 0, stream>>>(Xb, Wvb, bv,      Vt, 1, 1.0f);    // V transposed

  attn<<<dim3(S_LEN / 128, NH, BATCH), 256, 0, stream>>>(Qb, Kb, Vt, mask, out);
}
